// Encoder_62285615727083
// MI455X (gfx1250) — matrix operand order repaired and checked by reading, not compiled, not run on hardware
//
#include <hip/hip_runtime.h>
#include <cmath>

#define DIM   768
#define HEADS 12
#define DKH   64
#define SEQ   2048
#define BATCH 2
#define TOK   (BATCH * SEQ)
#define FF    (4 * DIM)

typedef __bf16 bf16_t;
typedef __attribute__((ext_vector_type(16))) __bf16 v16bf;
typedef __attribute__((ext_vector_type(8)))  __bf16 v8bf;
typedef __attribute__((ext_vector_type(8)))  float  v8f;

static __device__ inline v16bf cat8(v8bf a, v8bf b) {
  return __builtin_shufflevector(a, b, 0, 1, 2, 3, 4, 5, 6, 7, 8, 9, 10, 11, 12, 13, 14, 15);
}

// A-fragment (16x32 bf16): lane row m = lane%16; element e -> K = half*16 + e
static __device__ inline v16bf load_afrag(const bf16_t* rowptr, int hh) {
  v8bf r0 = *(const v8bf*)(rowptr + hh * 16);
  v8bf r1 = *(const v8bf*)(rowptr + hh * 16 + 8);
  return cat8(r0, r1);
}
// B-fragment (32x16 bf16): lane col n = lane%16; element e -> K = half*16 + e
static __device__ inline v16bf load_bfrag(const bf16_t* colptr) {
  v8bf r0 = *(const v8bf*)(colptr);
  v8bf r1 = *(const v8bf*)(colptr + 8);
  return cat8(r0, r1);
}

// ---- CDNA5 async copy: 16B per lane, global -> LDS, tracked by ASYNCcnt ----
static __device__ inline void async_cp16(const void* gptr, void* lptr) {
  unsigned lds = (unsigned)(uintptr_t)lptr;  // LDS aperture: low 32 bits = LDS offset
  asm volatile("global_load_async_to_lds_b128 %0, %1, off" ::"v"(lds), "v"(gptr)
               : "memory");
}
static __device__ inline void wait_async0() {
  asm volatile("s_wait_asynccnt 0" ::: "memory");
}
static __device__ inline void wait_ds0() {
  asm volatile("s_wait_dscnt 0" ::: "memory");
}

// ---- 16-lane butterfly reductions via ds_swizzle (xor masks 8,4,2,1) ----
#define SWZF(x, imm) __int_as_float(__builtin_amdgcn_ds_swizzle(__float_as_int(x), imm))
static __device__ inline float rmax16(float x) {
  x = fmaxf(x, SWZF(x, 0x201f));
  x = fmaxf(x, SWZF(x, 0x101f));
  x = fmaxf(x, SWZF(x, 0x081f));
  x = fmaxf(x, SWZF(x, 0x041f));
  return x;
}
static __device__ inline float rsum16(float x) {
  x += SWZF(x, 0x201f);
  x += SWZF(x, 0x101f);
  x += SWZF(x, 0x081f);
  x += SWZF(x, 0x041f);
  return x;
}

// ---------------------------------------------------------------------------
// f32 -> bf16 conversion (row-major, for activations)
// ---------------------------------------------------------------------------
__global__ void __launch_bounds__(256) f2bf_kernel(const float* __restrict__ in,
                                                   bf16_t* __restrict__ out, int n) {
  int i = blockIdx.x * blockDim.x + threadIdx.x;
  int stride = gridDim.x * blockDim.x;
  for (; i < n; i += stride) out[i] = (bf16_t)in[i];
}

// ---------------------------------------------------------------------------
// Weight transpose + convert: in[K][N] f32 -> out[N][K] bf16 (tiled via LDS)
// ---------------------------------------------------------------------------
__global__ void __launch_bounds__(256)
transpose_f2bf_kernel(const float* __restrict__ in, bf16_t* __restrict__ out,
                      int K, int N) {
  __shared__ float t[16][17];
  const int n0 = blockIdx.x * 16, k0 = blockIdx.y * 16;
  const int tx = threadIdx.x & 15, ty = threadIdx.x >> 4;
  t[ty][tx] = in[(size_t)(k0 + ty) * N + (n0 + tx)];
  __syncthreads();
  out[(size_t)(n0 + ty) * K + (k0 + tx)] = (bf16_t)t[tx][ty];
}

// ---------------------------------------------------------------------------
// Tiled bf16 GEMM with async double-buffered LDS staging.
//   C[M,N] = A[M,K] @ B[K,N] + bias     (B passed pre-transposed: Bt[N][K])
// Block: 128 threads = 4 waves; tile 64(M) x 64(N); K-step 32.
// MODE: 0 = f32 row-major out, 1 = bf16 row-major out,
//       2 = bf16 out in V-transposed layout [B,H,DKH,SEQ] (for attention V).
// ---------------------------------------------------------------------------
template <int MODE, int RELU>
__global__ void __launch_bounds__(128)
gemm_bf16_kernel(const bf16_t* __restrict__ A, const bf16_t* __restrict__ Bt,
                 const float* __restrict__ bias, void* __restrict__ Cout,
                 int M, int N, int K) {
  __shared__ alignas(16) bf16_t As[2][64][40];  // [m][k], padded stride 40
  __shared__ alignas(16) bf16_t Bs[2][64][40];  // [n][k], padded stride 40
  const int tid = threadIdx.x;
  const int wv = tid >> 5, lane = tid & 31;
  const int hh = lane >> 4, l16 = lane & 15;
  const int m0 = blockIdx.y * 64, n0 = blockIdx.x * 64;
  const int r = tid >> 1, cc = (tid & 1) * 16;
  v8f acc[4] = {};

  auto stage = [&](int k0, int p) {
    const bf16_t* a = A + (size_t)(m0 + r) * K + (k0 + cc);
    async_cp16(a, &As[p][r][cc]);
    async_cp16(a + 8, &As[p][r][cc + 8]);
    const bf16_t* bsrc = Bt + (size_t)(n0 + r) * K + (k0 + cc);
    async_cp16(bsrc, &Bs[p][r][cc]);
    async_cp16(bsrc + 8, &Bs[p][r][cc + 8]);
  };

  stage(0, 0);
  wait_async0();
  __syncthreads();

  int p = 0;
  for (int k0 = 0; k0 < K; k0 += 32) {
    if (k0 + 32 < K) stage(k0 + 32, p ^ 1);  // prefetch next tile (async)
    v16bf af = load_afrag(&As[p][wv * 16 + l16][0], hh);
#pragma unroll
    for (int nt = 0; nt < 4; ++nt) {
      v16bf bfr = load_bfrag(&Bs[p][nt * 16 + l16][hh * 16]);
      acc[nt] = __builtin_amdgcn_wmma_f32_16x16x32_bf16(false, af, false, bfr,
                                                        (short)0, acc[nt], false, false);
    }
    wait_ds0();     // our LDS reads done before buffer p is overwritten next iter
    wait_async0();  // our async writes into p^1 have landed
    __syncthreads();
    p ^= 1;
  }

#pragma unroll
  for (int nt = 0; nt < 4; ++nt) {
    const int col = n0 + nt * 16 + l16;
    const float bv = bias[col];
#pragma unroll
    for (int j = 0; j < 8; ++j) {
      const int row = m0 + wv * 16 + j + hh * 8;  // C/D: VGPR j -> rows j / j+8
      float v = acc[nt][j] + bv;
      if (RELU) v = fmaxf(v, 0.f);
      if (MODE == 0) {
        ((float*)Cout)[(size_t)row * N + col] = v;
      } else if (MODE == 1) {
        ((bf16_t*)Cout)[(size_t)row * N + col] = (bf16_t)v;
      } else {  // V-transposed: [B,H,DKH,SEQ]
        const int hd = col >> 6, d = col & 63;
        const int bb = row >> 11, s = row & (SEQ - 1);
        ((bf16_t*)Cout)[(((size_t)(bb * HEADS + hd)) * DKH + d) * SEQ + s] = (bf16_t)v;
      }
    }
  }
}

// ---------------------------------------------------------------------------
// Flash attention: block = (b, h, 64 q-rows), 4 waves x 16 q-rows.
// K streamed 32 keys/iter; K tiles + pre-transposed V tiles staged via async
// double-buffered LDS. Online softmax in fp32 with reference semantics:
// scale = 1/sqrt(768); masked score := 1e-11 (participates in softmax).
// ---------------------------------------------------------------------------
__global__ void __launch_bounds__(128)
attn_kernel(const bf16_t* __restrict__ Q, const bf16_t* __restrict__ Kg,
            const bf16_t* __restrict__ Vt, const int* __restrict__ mask,
            bf16_t* __restrict__ Z) {
  __shared__ alignas(16) bf16_t Ks[2][32][72];    // [key][feat 64], padded
  __shared__ alignas(16) bf16_t Vs[2][64][40];    // [feat][key 32], padded
  __shared__ alignas(16) bf16_t Ps[4][16][40];    // per-wave P tile
  const int tid = threadIdx.x;
  const int wv = tid >> 5, lane = tid & 31;
  const int hh = lane >> 4, l16 = lane & 15;
  const int qt4 = blockIdx.x & 31;
  const int bh = blockIdx.x >> 5;
  const int b = bh / HEADS, h = bh % HEADS;
  const int s0 = qt4 * 64 + wv * 16;
  const float scale = 0.036084391824351614f;  // 1/sqrt(768)
  const int kr = tid >> 2, kc = (tid & 3) * 16;  // K tile 32x64
  const int vr = tid >> 1, vc = (tid & 1) * 16;  // V tile 64x32

  auto stage = [&](int key0, int p) {
    const bf16_t* ks = Kg + ((size_t)(b * SEQ + key0 + kr)) * DIM + h * DKH + kc;
    async_cp16(ks, &Ks[p][kr][kc]);
    async_cp16(ks + 8, &Ks[p][kr][kc + 8]);
    const bf16_t* vs = Vt + ((size_t)(bh * DKH + vr)) * SEQ + key0 + vc;
    async_cp16(vs, &Vs[p][vr][vc]);
    async_cp16(vs + 8, &Vs[p][vr][vc + 8]);
  };

  v16bf qf[2];
#pragma unroll
  for (int c = 0; c < 2; ++c) {
    const bf16_t* qp = Q + ((size_t)(b * SEQ + s0 + l16)) * DIM + h * DKH + c * 32;
    qf[c] = load_afrag(qp, hh);
  }

  float mrow[8], lrow[8];
  v8f o[4] = {};
#pragma unroll
  for (int j = 0; j < 8; ++j) { mrow[j] = -3.0e38f; lrow[j] = 0.f; }

  stage(0, 0);
  wait_async0();
  __syncthreads();

  int p = 0;
  for (int key0 = 0; key0 < SEQ; key0 += 32) {
    if (key0 + 32 < SEQ) stage(key0 + 32, p ^ 1);  // async prefetch next K/V tile

    // scores S[16q x 32keys]: 2 key subtiles x 2 feature chunks
    v8f sc[2] = {};
#pragma unroll
    for (int t = 0; t < 2; ++t) {
#pragma unroll
      for (int c = 0; c < 2; ++c) {
        v16bf kf = load_bfrag(&Ks[p][t * 16 + l16][c * 32 + hh * 16]);
        sc[t] = __builtin_amdgcn_wmma_f32_16x16x32_bf16(false, qf[c], false, kf,
                                                        (short)0, sc[t], false, false);
      }
    }

    const int mv0 = mask[b * SEQ + key0 + l16];
    const int mv1 = mask[b * SEQ + key0 + 16 + l16];
#pragma unroll
    for (int j = 0; j < 8; ++j) {
      float v0 = (mv0 == 0) ? 1e-11f : sc[0][j] * scale;
      float v1 = (mv1 == 0) ? 1e-11f : sc[1][j] * scale;
      const float mnew = fmaxf(mrow[j], rmax16(fmaxf(v0, v1)));
      const float alpha = __expf(mrow[j] - mnew);
      const float p0 = __expf(v0 - mnew);
      const float p1 = __expf(v1 - mnew);
      lrow[j] = lrow[j] * alpha + rsum16(p0 + p1);
      mrow[j] = mnew;
      Ps[wv][j + hh * 8][l16] = (bf16_t)p0;  // C-frag -> A layout via LDS
      Ps[wv][j + hh * 8][16 + l16] = (bf16_t)p1;
#pragma unroll
      for (int nt = 0; nt < 4; ++nt) o[nt][j] *= alpha;
    }

    wait_ds0();  // wave-private P tile RAW guard
    v16bf pf = load_afrag(&Ps[wv][l16][0], hh);
#pragma unroll
    for (int nt = 0; nt < 4; ++nt) {
      v16bf vf = load_bfrag(&Vs[p][nt * 16 + l16][hh * 16]);
      o[nt] = __builtin_amdgcn_wmma_f32_16x16x32_bf16(false, pf, false, vf,
                                                      (short)0, o[nt], false, false);
    }

    wait_ds0();     // our LDS reads done before buffer p overwritten next iter
    wait_async0();  // next tile fully landed
    __syncthreads();
    p ^= 1;
  }

#pragma unroll
  for (int nt = 0; nt < 4; ++nt) {
#pragma unroll
    for (int j = 0; j < 8; ++j) {
      const int row = s0 + j + hh * 8;
      const int col = h * DKH + nt * 16 + l16;
      Z[((size_t)(b * SEQ + row)) * DIM + col] = (bf16_t)(o[nt][j] / lrow[j]);
    }
  }
}

// ---------------------------------------------------------------------------
// out = resid + LayerNorm(v)*g + b ; optionally also emit bf16 copy.
// ---------------------------------------------------------------------------
__global__ void __launch_bounds__(256)
addnorm_kernel(const float* __restrict__ resid, const float* __restrict__ v,
               const float* __restrict__ g, const float* __restrict__ bta,
               float* __restrict__ out, bf16_t* __restrict__ outb) {
  const int row = blockIdx.x;
  const int t = threadIdx.x;
  const int wv = t >> 5, lane = t & 31;
  const float* x = v + (size_t)row * DIM;
  __shared__ float sh1[8], sh2[8];
  float loc[3];
  float s = 0.f;
#pragma unroll
  for (int i = 0; i < 3; ++i) { loc[i] = x[t + i * 256]; s += loc[i]; }
#pragma unroll
  for (int m = 16; m >= 1; m >>= 1) s += __shfl_xor(s, m, 32);
  if (lane == 0) sh1[wv] = s;
  __syncthreads();
  float tot = 0.f;
#pragma unroll
  for (int i = 0; i < 8; ++i) tot += sh1[i];
  const float mu = tot * (1.0f / DIM);
  float s2 = 0.f;
#pragma unroll
  for (int i = 0; i < 3; ++i) { float d = loc[i] - mu; s2 += d * d; }
#pragma unroll
  for (int m = 16; m >= 1; m >>= 1) s2 += __shfl_xor(s2, m, 32);
  if (lane == 0) sh2[wv] = s2;
  __syncthreads();
  float tot2 = 0.f;
#pragma unroll
  for (int i = 0; i < 8; ++i) tot2 += sh2[i];
  const float rs = rsqrtf(tot2 * (1.0f / DIM) + 1e-5f);
#pragma unroll
  for (int i = 0; i < 3; ++i) {
    const int col = t + i * 256;
    const float y = resid[(size_t)row * DIM + col] + (loc[i] - mu) * rs * g[col] + bta[col];
    out[(size_t)row * DIM + col] = y;
    if (outb) outb[(size_t)row * DIM + col] = (bf16_t)y;
  }
}

// ---------------------------------------------------------------------------
extern "C" void kernel_launch(void* const* d_in, const int* in_sizes, int n_in,
                              void* d_out, int out_size, void* d_ws, size_t ws_size,
                              hipStream_t stream) {
  const float* x_n = (const float*)d_in[0];
  const int* mask = (const int*)d_in[1];
  const float* wq = (const float*)d_in[2];
  const float* bq = (const float*)d_in[3];
  const float* wk = (const float*)d_in[4];
  const float* bk = (const float*)d_in[5];
  const float* wvp = (const float*)d_in[6];
  const float* bv = (const float*)d_in[7];
  const float* wo = (const float*)d_in[8];
  const float* bo = (const float*)d_in[9];
  const float* ln1_g = (const float*)d_in[10];
  const float* ln1_b = (const float*)d_in[11];
  const float* w1 = (const float*)d_in[12];
  const float* b1 = (const float*)d_in[13];
  const float* w2 = (const float*)d_in[14];
  const float* b2 = (const float*)d_in[15];
  const float* ln2_g = (const float*)d_in[16];
  const float* ln2_b = (const float*)d_in[17];

  char* ws = (char*)d_ws;
  size_t off = 0;
  auto take = [&](size_t bytes) -> void* {
    void* p = ws + off;
    off += (bytes + 255) & ~(size_t)255;
    return p;
  };
  bf16_t* xb  = (bf16_t*)take((size_t)TOK * DIM * 2);
  bf16_t* wqt = (bf16_t*)take((size_t)DIM * DIM * 2);
  bf16_t* wkt = (bf16_t*)take((size_t)DIM * DIM * 2);
  bf16_t* wvt = (bf16_t*)take((size_t)DIM * DIM * 2);
  bf16_t* wot = (bf16_t*)take((size_t)DIM * DIM * 2);
  bf16_t* w1t = (bf16_t*)take((size_t)DIM * FF * 2);
  bf16_t* w2t = (bf16_t*)take((size_t)FF * DIM * 2);
  bf16_t* qb  = (bf16_t*)take((size_t)TOK * DIM * 2);
  bf16_t* kb  = (bf16_t*)take((size_t)TOK * DIM * 2);
  bf16_t* vtb = (bf16_t*)take((size_t)TOK * DIM * 2);  // [B,H,DKH,SEQ]
  bf16_t* zb  = (bf16_t*)take((size_t)TOK * DIM * 2);
  bf16_t* l1b = (bf16_t*)take((size_t)TOK * DIM * 2);
  bf16_t* hb  = (bf16_t*)take((size_t)TOK * FF * 2);
  float*  tbuf = (float*)take((size_t)TOK * DIM * 4);  // attn-proj, later FFN2 out
  float*  l1   = (float*)take((size_t)TOK * DIM * 4);

  f2bf_kernel<<<512, 256, 0, stream>>>(x_n, xb, TOK * DIM);
  auto tr = [&](const float* src, bf16_t* dst, int K, int N) {
    transpose_f2bf_kernel<<<dim3(N / 16, K / 16), 256, 0, stream>>>(src, dst, K, N);
  };
  tr(wq, wqt, DIM, DIM);
  tr(wk, wkt, DIM, DIM);
  tr(wvp, wvt, DIM, DIM);
  tr(wo, wot, DIM, DIM);
  tr(w1, w1t, DIM, FF);
  tr(w2, w2t, FF, DIM);

  dim3 gsq(DIM / 64, TOK / 64);
  gemm_bf16_kernel<1, 0><<<gsq, 128, 0, stream>>>(xb, wqt, bq, qb, TOK, DIM, DIM);
  gemm_bf16_kernel<1, 0><<<gsq, 128, 0, stream>>>(xb, wkt, bk, kb, TOK, DIM, DIM);
  gemm_bf16_kernel<2, 0><<<gsq, 128, 0, stream>>>(xb, wvt, bv, vtb, TOK, DIM, DIM);

  attn_kernel<<<BATCH * HEADS * (SEQ / 64), 128, 0, stream>>>(qb, kb, vtb, mask, zb);

  gemm_bf16_kernel<0, 0><<<gsq, 128, 0, stream>>>(zb, wot, bo, tbuf, TOK, DIM, DIM);
  addnorm_kernel<<<TOK, 256, 0, stream>>>(x_n, tbuf, ln1_g, ln1_b, l1, l1b);

  dim3 gff1(FF / 64, TOK / 64);
  gemm_bf16_kernel<1, 1><<<gff1, 128, 0, stream>>>(l1b, w1t, b1, hb, TOK, FF, DIM);
  gemm_bf16_kernel<0, 0><<<gsq, 128, 0, stream>>>(hb, w2t, b2, tbuf, TOK, DIM, FF);
  addnorm_kernel<<<TOK, 256, 0, stream>>>(l1, tbuf, ln2_g, ln2_b, (float*)d_out, nullptr);
}
